// MatMulRemapper_22574348108404
// MI455X (gfx1250) — matrix operand order repaired and checked by reading, not compiled, not run on hardware
//
#include <hip/hip_runtime.h>
#include <hip/hip_bf16.h>

// Batched GEMM: out[g] = X[g](256x4096) * B[g](4096x4096), fp32 in/out,
// bf16 WMMA core (f32 accumulate). Memory-bound by the 537MB interp matrix.

#define NB 8
#define M  256
#define K  4096
#define NN 4096
#define BM 128
#define BN 128
#define BK 32
#define KT (K / BK)   // 128 k-tiles
#define LDT 56        // LDS row stride in bf16 elems (112B: 16B aligned, bank-friendly)

typedef __attribute__((ext_vector_type(16))) __bf16       v16bf;
typedef __attribute__((ext_vector_type(4)))  __bf16       bf16x4;
typedef __attribute__((ext_vector_type(2)))  __bf16       bf16x2;
typedef __attribute__((ext_vector_type(8)))  float        v8f;
typedef __attribute__((ext_vector_type(4)))  unsigned int u32x4;

union Frag { v16bf bf; u32x4 u[2]; };

__device__ __forceinline__ bf16x2 cvt2(float a, float b) {
  bf16x2 r; r.x = (__bf16)a; r.y = (__bf16)b; return r;   // native v_cvt path, RNE
}

__global__ __launch_bounds__(256)
void remap_gemm_bf16(const float* __restrict__ X,
                     const float* __restrict__ Bmat,
                     float* __restrict__ Out) {
  __shared__ __align__(16) unsigned short sA[2][BM * LDT]; // [row m][k]  bf16
  __shared__ __align__(16) unsigned short sB[2][BN * LDT]; // [col n][k]  bf16 (B transposed)

  const int g  = blockIdx.z;
  const int n0 = blockIdx.x * BN;
  const int m0 = blockIdx.y * BM;
  const float* Ag = X    + (size_t)g * M * K;
  const float* Bg = Bmat + (size_t)g * K * NN;
  float*       Cg = Out  + (size_t)g * M * NN;

  const int tid  = threadIdx.x;
  const int lane = tid & 31;          // wave32
  const int wave = tid >> 5;          // 8 waves
  const int wm   = wave >> 2;         // 0..1 -> 64 rows each
  const int wn   = wave & 3;          // 0..3 -> 32 cols each
  const int l16  = lane & 15;
  const int hi   = lane >> 4;

  float4 ar[4], br[4];                // staging registers (global -> LDS pipeline)

  // A tile: 128 rows x 32 k, one float4 per thread per pass
  const int a_row = tid >> 3;                 // 0..31 (+32 per pass)
  const int a_c4  = (tid & 7) * 4;            // 0..28
  // B tile: 32 k x 128 n, thread handles k-pair rows (kp, kp+1), 4 columns
  const int b_kp0 = (tid >> 5) * 2;           // pass 0: 0..14 even
  const int b_kp1 = ((256 + tid) >> 5) * 2 - 16; // pass 1 offset handled below
  const int b_c4  = (tid & 31) * 4;           // 0..124

  auto loadA = [&](int kt) {
    const int kk = kt * BK;
    #pragma unroll
    for (int p = 0; p < 4; ++p)
      ar[p] = *(const float4*)(Ag + (size_t)(m0 + p * 32 + a_row) * K + kk + a_c4);
  };
  auto loadB = [&](int kt) {
    const int kk = kt * BK;
    #pragma unroll
    for (int p = 0; p < 2; ++p) {
      const int kp = p * 16 + b_kp0;          // 0..30 even
      br[2 * p + 0] = *(const float4*)(Bg + (size_t)(kk + kp + 0) * NN + n0 + b_c4);
      br[2 * p + 1] = *(const float4*)(Bg + (size_t)(kk + kp + 1) * NN + n0 + b_c4);
    }
  };
  auto storeA = [&](int buf) {                // f32 -> bf16, row-major [m][k]
    #pragma unroll
    for (int p = 0; p < 4; ++p) {
      bf16x4 v;
      v.x = (__bf16)ar[p].x; v.y = (__bf16)ar[p].y;
      v.z = (__bf16)ar[p].z; v.w = (__bf16)ar[p].w;
      *(bf16x4*)&sA[buf][(p * 32 + a_row) * LDT + a_c4] = v;
    }
  };
  auto storeB = [&](int buf) {                // f32 -> bf16, transpose to [n][k], k-pairs packed
    #pragma unroll
    for (int p = 0; p < 2; ++p) {
      const int kp = p * 16 + b_kp0;
      const float4 r0 = br[2 * p + 0], r1 = br[2 * p + 1];
      *(bf16x2*)&sB[buf][(b_c4 + 0) * LDT + kp] = cvt2(r0.x, r1.x);
      *(bf16x2*)&sB[buf][(b_c4 + 1) * LDT + kp] = cvt2(r0.y, r1.y);
      *(bf16x2*)&sB[buf][(b_c4 + 2) * LDT + kp] = cvt2(r0.z, r1.z);
      *(bf16x2*)&sB[buf][(b_c4 + 3) * LDT + kp] = cvt2(r0.w, r1.w);
    }
  };

  v8f acc[4][2] = {};

  auto compute = [&](int buf) {
    Frag fa[4], fb[2];
    // A fragment, 16x32 bf16 (lane<16 -> K 0..7 & 16..23; lane>=16 -> K 8..15 & 24..31)
    #pragma unroll
    for (int i = 0; i < 4; ++i) {
      const unsigned short* base = &sA[buf][(wm * 64 + i * 16 + l16) * LDT + hi * 8];
      fa[i].u[0] = *(const u32x4*)(base);
      fa[i].u[1] = *(const u32x4*)(base + 16);
    }
    // B fragment, 32x16 bf16 (lane<16 -> K 0..7 & 16..23 of col N=lane; lane>=16 -> K 8..15 & 24..31)
    #pragma unroll
    for (int j = 0; j < 2; ++j) {
      const unsigned short* base = &sB[buf][(wn * 32 + j * 16 + l16) * LDT + hi * 8];
      fb[j].u[0] = *(const u32x4*)(base);
      fb[j].u[1] = *(const u32x4*)(base + 16);
    }
    #pragma unroll
    for (int i = 0; i < 4; ++i)
      #pragma unroll
      for (int j = 0; j < 2; ++j)
        acc[i][j] = __builtin_amdgcn_wmma_f32_16x16x32_bf16(
            false, fa[i].bf, false, fb[j].bf, (short)0, acc[i][j], false, false);
  };

  // prologue: stage k-tile 0
  loadA(0); loadB(0);
  storeA(0); storeB(0);
  __syncthreads();

  for (int kt = 0; kt < KT; ++kt) {
    const int buf = kt & 1;
    if (kt + 1 < KT) { loadA(kt + 1); loadB(kt + 1); }   // overlap with compute
    if (kt + 2 < KT) {                                    // prefetch the streamed B matrix
      const int kk2 = (kt + 2) * BK;
      __builtin_prefetch(Bg + (size_t)(kk2 + b_kp0) * NN + n0 + b_c4, 0, 1);
    }
    compute(buf);
    if (kt + 1 < KT) { storeA(buf ^ 1); storeB(buf ^ 1); }
    __syncthreads();
  }

  // C/D 16x16 f32 layout: VGPR r -> M = r (lanes 0-15) / M = 8+r (lanes 16-31), N = lane&15
  #pragma unroll
  for (int i = 0; i < 4; ++i) {
    #pragma unroll
    for (int j = 0; j < 2; ++j) {
      const int n  = n0 + wn * 32 + j * 16 + l16;
      const int mb = m0 + wm * 64 + i * 16 + hi * 8;
      #pragma unroll
      for (int r = 0; r < 8; ++r)
        Cg[(size_t)(mb + r) * NN + n] = acc[i][j][r];
    }
  }
}

extern "C" void kernel_launch(void* const* d_in, const int* in_sizes, int n_in,
                              void* d_out, int out_size, void* d_ws, size_t ws_size,
                              hipStream_t stream) {
  const float* x = (const float*)d_in[0];          // [8,256,64,64] f32
  const float* m = (const float*)d_in[1];          // [8,4096,4096] f32 (interp_mat_t)
  float* out = (float*)d_out;                      // [8,256,64,64] f32
  dim3 grid(NN / BN, M / BM, NB);                  // 32 x 2 x 8 = 512 blocks
  remap_gemm_bf16<<<grid, dim3(256), 0, stream>>>(x, m, out);
}
